// CrossAttention_20057497272652
// MI455X (gfx1250) — matrix operand order repaired and checked by reading, not compiled, not run on hardware
//
#include <hip/hip_runtime.h>
#include <hip/hip_bf16.h>

// ---------------------------------------------------------------------------
// CDNA5 (gfx1250) fused cross-attention: bf16 WMMA + TDM tensor_load_to_lds.
// N=2, T=1024, S=2048, D=1024, H=16, DH=64
// ---------------------------------------------------------------------------

typedef __attribute__((ext_vector_type(16))) __bf16 v16bf;
typedef __attribute__((ext_vector_type(8)))  float  v8f;
typedef __attribute__((ext_vector_type(4)))  unsigned int v4u;
typedef __attribute__((ext_vector_type(8)))  int    v8i;
typedef __attribute__((ext_vector_type(4)))  int    v4i;

#define NB 2
#define TT 1024
#define SS 2048
#define DM 1024
#define HH 16
#define DH 64

static __device__ __forceinline__ v8f wmma_bf16(v16bf a, v16bf b, v8f c) {
  return __builtin_amdgcn_wmma_f32_16x16x32_bf16(false, a, false, b, (short)0, c,
                                                 false, false);
}

// A fragment: 16x32 bf16, row-major, row stride lda. Per ISA 7.12.2:
// lanes 0-15 (M=lane) hold K {0..7,16..23}; lanes 16-31 hold K {8..15,24..31};
// VGPR j packs two consecutive K values -> 2x16B contiguous per lane.
static __device__ __forceinline__ v16bf load_a_frag(const __bf16* __restrict__ p,
                                                    int lda) {
  int lane = threadIdx.x & 31;
  int row  = lane & 15;
  int ks   = lane >> 4;
  const __bf16* r = p + (size_t)row * lda + ks * 8;
  v16bf a;
#pragma unroll
  for (int i = 0; i < 8; ++i) {
    int k = ((i & 3) * 2) + ((i >> 2) * 16);
    a[2 * i]     = r[k];
    a[2 * i + 1] = r[k + 1];
  }
  return a;
}

// A fragment built from f32 data (LDS attention probabilities).
static __device__ __forceinline__ v16bf load_a_frag_f32(const float* __restrict__ p,
                                                        int lda) {
  int lane = threadIdx.x & 31;
  int row  = lane & 15;
  int ks   = lane >> 4;
  const float* r = p + (size_t)row * lda + ks * 8;
  v16bf a;
#pragma unroll
  for (int i = 0; i < 8; ++i) {
    int k = ((i & 3) * 2) + ((i >> 2) * 16);
    a[2 * i]     = (__bf16)r[k];
    a[2 * i + 1] = (__bf16)r[k + 1];
  }
  return a;
}

// B fragment where element (k,col) lives at p[col*stride + k] (k contiguous).
// Lanes 0-15: N=col, K {0..7,16..23}; lanes 16-31: K {8..15,24..31}. 2x16B per lane.
static __device__ __forceinline__ v16bf load_bt_frag(const __bf16* __restrict__ p,
                                                     int stride) {
  int lane = threadIdx.x & 31;
  int col  = lane & 15;
  int ks   = lane >> 4;
  const __bf16* c = p + (size_t)col * stride + ks * 8;
  v16bf b;
#pragma unroll
  for (int i = 0; i < 8; ++i) {
    int k = ((i & 3) * 2) + ((i >> 2) * 16);
    b[2 * i]     = c[k];
    b[2 * i + 1] = c[k + 1];
  }
  return b;
}

// ---------------------------------------------------------------------------
// f32 -> bf16 elementwise conversion
// ---------------------------------------------------------------------------
__global__ void cvt_f32_bf16(const float* __restrict__ in, __bf16* __restrict__ out,
                             int n) {
  int i = blockIdx.x * blockDim.x + threadIdx.x;
  if (i < n) out[i] = (__bf16)in[i];
}

// ---------------------------------------------------------------------------
// f32 -> bf16 transpose (dim x dim), LDS tiled 32x32. out[c*dim+r] = in[r*dim+c]
// ---------------------------------------------------------------------------
__global__ __launch_bounds__(256) void transpose_f32_bf16(
    const float* __restrict__ in, __bf16* __restrict__ out, int dim) {
  __shared__ float tile[32][33];
  int bx = blockIdx.x * 32, by = blockIdx.y * 32;
  int tx = threadIdx.x & 31, ty = threadIdx.x >> 5;   // 32 x 8
#pragma unroll
  for (int r = ty; r < 32; r += 8)
    tile[r][tx] = in[(size_t)(by + r) * dim + bx + tx];
  __syncthreads();
#pragma unroll
  for (int r = ty; r < 32; r += 8)
    out[(size_t)(bx + r) * dim + by + tx] = (__bf16)tile[tx][r];
}

// ---------------------------------------------------------------------------
// GEMM: C[M,1024] = A(bf16)[M,1024] x W + bias, with W given TRANSPOSED
// (Bt[n][k], n-major) so B fragments are two contiguous 16B runs per lane.
// Block 256 = 8 waves; wave tile 16x64; WG tile 128x64. grid=(M/128, 16).
// Output: exactly one of outBf ([M,D] bf16), outF ([M,D] f32),
// outVt (per-head transposed bf16 [n][h][dh][key]) is non-null.
// ---------------------------------------------------------------------------
__global__ __launch_bounds__(256) void gemm_bf16(
    const __bf16* __restrict__ A, const __bf16* __restrict__ Bt,
    const float* __restrict__ bias, __bf16* __restrict__ outBf,
    float* __restrict__ outF, __bf16* __restrict__ outVt) {
  const int K = DM;
  int wave = threadIdx.x >> 5;
  int lane = threadIdx.x & 31;
  int rowBase = blockIdx.x * 128 + wave * 16;
  int colBase = blockIdx.y * 64;

  v8f acc[4];
#pragma unroll
  for (int t = 0; t < 4; ++t) acc[t] = (v8f){};

  for (int k0 = 0; k0 < K; k0 += 32) {
    const __bf16* ap = A + (size_t)rowBase * K + k0;
    __builtin_prefetch(ap + 64, 0, 1);   // stream-ahead on A
    v16bf a = load_a_frag(ap, K);
#pragma unroll
    for (int t = 0; t < 4; ++t) {
      v16bf b = load_bt_frag(Bt + (size_t)(colBase + t * 16) * K + k0, K);
      acc[t] = wmma_bf16(a, b, acc[t]);
    }
  }

  int ks   = lane >> 4;
  int col0 = lane & 15;
#pragma unroll
  for (int t = 0; t < 4; ++t) {
#pragma unroll
    for (int i = 0; i < 8; ++i) {
      int row = rowBase + i + 8 * ks;
      int col = colBase + t * 16 + col0;
      float v = acc[t][i] + bias[col];
      if (outF) {
        outF[(size_t)row * DM + col] = v;
      } else if (outVt) {
        int nb = row / SS, key = row % SS;
        int h = col / DH, dh = col % DH;
        outVt[(((size_t)nb * HH + h) * DH + dh) * SS + key] = (__bf16)v;
      } else {
        outBf[(size_t)row * DM + col] = (__bf16)v;
      }
    }
  }
}

// ---------------------------------------------------------------------------
// Fused attention per (n, h, 16-query tile).
// Phase 0: TDM tensor_load_to_lds stages the 16x64 bf16 Q tile into LDS
// Phase 1: scores = QK^T * scale (+mask) -> 128 KB LDS
// Phase 2: row softmax (wave32 shuffle reductions), write attn f32 to d_out
// Phase 3: AV (V^T layout, contiguous keys), split-K over waves, LDS reduce
// ---------------------------------------------------------------------------
__global__ __launch_bounds__(256) void attn_fused(
    const __bf16* __restrict__ Qb, const __bf16* __restrict__ Kb,
    const __bf16* __restrict__ Vt, const unsigned char* __restrict__ mask,
    float* __restrict__ attnOut, __bf16* __restrict__ Ob) {
  extern __shared__ char smem[];
  __bf16* qtile = (__bf16*)smem;                       // 16x64 bf16 = 2 KB
  float*  sc    = (float*)(smem + 2048);               // [16][SS]   = 128 KB
  float*  part  = (float*)(smem + 2048 + 16 * SS * 4); // [4][256]   = 4 KB

  const int QT = TT / 16;
  int bx = blockIdx.x;
  int qt = bx % QT;
  int h  = (bx / QT) % HH;
  int n  = bx / (QT * HH);
  int qbase = qt * 16;

  int wave = threadIdx.x >> 5;
  int lane = threadIdx.x & 31;
  int col0 = lane & 15;
  int ks   = lane >> 4;
  const float scale = 0.125f;

  // ---- Phase 0: TDM async tensor load of Q tile (16 rows x 64 cols) ------
  if (wave == 0) {
    unsigned long long ga =
        (unsigned long long)(uintptr_t)(Qb + ((size_t)(n * TT) + qbase) * DM + h * DH);
    unsigned int ldsOff = (unsigned int)(uintptr_t)qtile;  // LDS byte offset
    v4u g0 = (v4u){};
    g0[0] = 1u;                                   // count=1 (valid D#)
    g0[1] = ldsOff;                               // lds_addr
    g0[2] = (unsigned int)(ga & 0xFFFFFFFFu);     // global_addr[31:0]
    g0[3] = (unsigned int)((ga >> 32) & 0x01FFFFFFu) | (2u << 30);  // [56:32], type=2
    v8i g1 = (v8i){};
    g1[0] = (int)(1u << 16);                      // data_size=1 (2 bytes/elem)
    g1[1] = (int)((DM & 0xFFFFu) << 16);          // tensor_dim0 = 1024 (lo16)
    g1[2] = (int)(((DM >> 16) & 0xFFFFu) |        // tensor_dim0 hi16
                  ((4096u & 0xFFFFu) << 16));     // tensor_dim1 = 4096 (lo16)
    g1[3] = (int)(((4096u >> 16) & 0xFFFFu) |     // tensor_dim1 hi16
                  (64u << 16));                   // tile_dim0 = 64
    g1[4] = 16;                                   // tile_dim1 = 16, tile_dim2 = 0
    g1[5] = DM;                                   // tensor_dim0_stride = 1024 (lo32)
    g1[6] = 0;
    g1[7] = 0;
    v4i g2 = (v4i){};
    v4i g3 = (v4i){};
    v8i g4 = (v8i){};
    __builtin_amdgcn_tensor_load_to_lds(g0, g1, g2, g3, g4, 0);
    __builtin_amdgcn_s_wait_tensorcnt(0);
  }
  __syncthreads();

  // ---- Phase 1: scores -> LDS --------------------------------------------
  v16bf a0 = load_a_frag(qtile, DH);
  v16bf a1 = load_a_frag(qtile + 32, DH);

  for (int kt = wave; kt < SS / 16; kt += 8) {
    const __bf16* kp = Kb + ((size_t)(n * SS) + kt * 16) * DM + h * DH;
    v8f acc = (v8f){};
    acc = wmma_bf16(a0, load_bt_frag(kp, DM), acc);
    acc = wmma_bf16(a1, load_bt_frag(kp + 32, DM), acc);
#pragma unroll
    for (int i = 0; i < 8; ++i) {
      int r   = i + 8 * ks;
      int key = kt * 16 + col0;
      unsigned char mv = mask[((size_t)(n * TT) + qbase + r) * SS + key];
      sc[r * SS + key] = acc[i] * scale + (mv ? 0.0f : -1e30f);
    }
  }
  __syncthreads();

  // ---- Phase 2: softmax over S, write attn to global ---------------------
#pragma unroll
  for (int rr = 0; rr < 2; ++rr) {
    int r = wave * 2 + rr;
    float m = -1e30f;
    for (int c = lane; c < SS; c += 32) m = fmaxf(m, sc[r * SS + c]);
#pragma unroll
    for (int off = 16; off; off >>= 1) m = fmaxf(m, __shfl_xor(m, off, 32));
    float sum = 0.0f;
    for (int c = lane; c < SS; c += 32) {
      float e = __expf(sc[r * SS + c] - m);
      sc[r * SS + c] = e;
      sum += e;
    }
#pragma unroll
    for (int off = 16; off; off >>= 1) sum += __shfl_xor(sum, off, 32);
    float inv = 1.0f / sum;
    float* arow = attnOut + ((size_t)((n * HH + h) * TT) + qbase + r) * SS;
    for (int c = lane; c < SS; c += 32) {
      float p = sc[r * SS + c] * inv;
      sc[r * SS + c] = p;
      arow[c] = p;
    }
  }
  __syncthreads();

  // ---- Phase 3: out = attn @ V, V^T layout so keys are contiguous --------
  int tn = wave & 3;     // 16-wide block of the 64-wide head dim
  int kh = wave >> 2;    // key half
  const __bf16* vp = Vt + (((size_t)n * HH + h) * DH + tn * 16) * SS;
  v8f acc = (v8f){};
  for (int k0 = kh * 1024; k0 < kh * 1024 + 1024; k0 += 32) {
    v16bf a = load_a_frag_f32(sc + k0, SS);
    v16bf b = load_bt_frag(vp + k0, SS);
    acc = wmma_bf16(a, b, acc);
  }
  if (kh == 1) {
#pragma unroll
    for (int i = 0; i < 8; ++i) part[tn * 256 + i * 32 + lane] = acc[i];
  }
  __syncthreads();
  if (kh == 0) {
#pragma unroll
    for (int i = 0; i < 8; ++i) {
      float v = acc[i] + part[tn * 256 + i * 32 + lane];
      int row = qbase + i + 8 * ks;
      int col = h * DH + tn * 16 + col0;
      Ob[((size_t)(n * TT) + row) * DM + col] = (__bf16)v;
    }
  }
}

// ---------------------------------------------------------------------------
// Host-side launch
// ---------------------------------------------------------------------------
extern "C" void kernel_launch(void* const* d_in, const int* in_sizes, int n_in,
                              void* d_out, int out_size, void* d_ws, size_t ws_size,
                              hipStream_t stream) {
  const float* target = (const float*)d_in[0];
  const float* source = (const float*)d_in[1];
  const unsigned char* mask = (const unsigned char*)d_in[2];
  const float* Wq = (const float*)d_in[3];
  const float* bq = (const float*)d_in[4];
  const float* Wk = (const float*)d_in[5];
  const float* bk = (const float*)d_in[6];
  const float* Wv = (const float*)d_in[7];
  const float* bv = (const float*)d_in[8];
  const float* Wo = (const float*)d_in[9];
  const float* bo = (const float*)d_in[10];

  float* out     = (float*)d_out;
  float* attnOut = out + (size_t)NB * TT * DM;

  char* ws = (char*)d_ws;
  const size_t SZ_T = (size_t)NB * TT * DM;   // 2M elems
  const size_t SZ_S = (size_t)NB * SS * DM;   // 4M elems
  const size_t SZ_W = (size_t)DM * DM;        // 1M elems
  __bf16* tgt_bf = (__bf16*)ws;  ws += SZ_T * 2;
  __bf16* src_bf = (__bf16*)ws;  ws += SZ_S * 2;
  __bf16* wqT    = (__bf16*)ws;  ws += SZ_W * 2;   // transposed weights
  __bf16* wkT    = (__bf16*)ws;  ws += SZ_W * 2;
  __bf16* wvT    = (__bf16*)ws;  ws += SZ_W * 2;
  __bf16* woT    = (__bf16*)ws;  ws += SZ_W * 2;
  __bf16* Qb     = (__bf16*)ws;  ws += SZ_T * 2;
  __bf16* Kb     = (__bf16*)ws;  ws += SZ_S * 2;
  __bf16* Vt     = (__bf16*)ws;  ws += SZ_S * 2;   // [n][h][dh][key]
  __bf16* Ob     = (__bf16*)ws;  ws += SZ_T * 2;

  // ---- conversions / weight transposes -----------------------------------
  cvt_f32_bf16<<<dim3((unsigned)((SZ_T + 255) / 256)), 256, 0, stream>>>(target, tgt_bf, (int)SZ_T);
  cvt_f32_bf16<<<dim3((unsigned)((SZ_S + 255) / 256)), 256, 0, stream>>>(source, src_bf, (int)SZ_S);
  dim3 tg(DM / 32, DM / 32);
  transpose_f32_bf16<<<tg, 256, 0, stream>>>(Wq, wqT, DM);
  transpose_f32_bf16<<<tg, 256, 0, stream>>>(Wk, wkT, DM);
  transpose_f32_bf16<<<tg, 256, 0, stream>>>(Wv, wvT, DM);
  transpose_f32_bf16<<<tg, 256, 0, stream>>>(Wo, woT, DM);

  // ---- QKV projections ---------------------------------------------------
  gemm_bf16<<<dim3(TT * NB / 128, DM / 64), 256, 0, stream>>>(tgt_bf, wqT, bq, Qb, nullptr, nullptr);
  gemm_bf16<<<dim3(SS * NB / 128, DM / 64), 256, 0, stream>>>(src_bf, wkT, bk, Kb, nullptr, nullptr);
  gemm_bf16<<<dim3(SS * NB / 128, DM / 64), 256, 0, stream>>>(src_bf, wvT, bv, nullptr, nullptr, Vt);

  // ---- fused attention (2KB Q + 128KB scores + 4KB partials LDS) ---------
  const int shmem = 2048 + 16 * SS * 4 + 4 * 256 * 4;
  (void)hipFuncSetAttribute((const void*)attn_fused,
                            hipFuncAttributeMaxDynamicSharedMemorySize, shmem);
  attn_fused<<<dim3(NB * HH * (TT / 16)), 256, shmem, stream>>>(Qb, Kb, Vt, mask,
                                                                attnOut, Ob);

  // ---- output projection (f32 result) ------------------------------------
  gemm_bf16<<<dim3(TT * NB / 128, DM / 64), 256, 0, stream>>>(Ob, woT, bo, nullptr, out, nullptr);
}
